// CausalSelfAttention_61718680044355
// MI455X (gfx1250) — matrix operand order repaired and checked by reading, not compiled, not run on hardware
//
#include <hip/hip_runtime.h>
#include <stdint.h>

typedef unsigned short u16;
typedef float v8f __attribute__((ext_vector_type(8)));
typedef __bf16 v16bf __attribute__((ext_vector_type(16)));
typedef unsigned short v8u __attribute__((ext_vector_type(8)));
typedef unsigned short v16u __attribute__((ext_vector_type(16)));
typedef unsigned int u32x4 __attribute__((ext_vector_type(4)));
typedef int i32x4 __attribute__((ext_vector_type(4)));
typedef int i32x8 __attribute__((ext_vector_type(8)));

#define DEVINL __device__ __forceinline__

#if defined(__has_builtin)
#if __has_builtin(__builtin_amdgcn_tensor_load_to_lds)
#define HAVE_TDM 1
#endif
#endif
#ifndef HAVE_TDM
#define HAVE_TDM 0
#endif

static constexpr int BB = 2;
static constexpr int TT = 4096;
static constexpr int CC = 768;
static constexpr int HH = 12;
static constexpr int HD = 64;
static constexpr int C3 = 3 * CC;   // 2304

// round-to-nearest-even f32 -> bf16
DEVINL u16 f2bf(float f) {
  union { float f; unsigned u; } v; v.f = f;
  unsigned r = v.u + 0x7FFFu + ((v.u >> 16) & 1u);
  return (u16)(r >> 16);
}

DEVINL v16bf mk16(v8u lo, v8u hi) {
  v16u t = __builtin_shufflevector(lo, hi, 0,1,2,3,4,5,6,7,8,9,10,11,12,13,14,15);
  return __builtin_bit_cast(v16bf, t);
}

// A-matrix (16x32 bf16) fragment: lane holds row m; K halves at {g*8.., 16+g*8..}
DEVINL v16bf fragA(const u16* rowp, int kk, int g) {
  v8u lo = *(const v8u*)(rowp + kk + g * 8);
  v8u hi = *(const v8u*)(rowp + kk + 16 + g * 8);
  return mk16(lo, hi);
}
// B-matrix (32x16 bf16) fragment from N-major LDS tile: lane = col n; K halves at {g*8.., 16+g*8..}
DEVINL v16bf fragB(const u16* rowp, int kk, int g) {
  v8u lo = *(const v8u*)(rowp + kk + g * 8);
  v8u hi = *(const v8u*)(rowp + kk + 16 + g * 8);
  return mk16(lo, hi);
}

DEVINL v8f wmma_bf16(v16bf a, v16bf b, v8f c) {
  return __builtin_amdgcn_wmma_f32_16x16x32_bf16(false, a, false, b, (short)0, c, false, false);
}

#if HAVE_TDM
// Issue a 2D TENSOR_LOAD_TO_LDS per CDNA5 D# layout (ISA cdna5 ch.8).
// data_size = 4B; tile_w_dw = tile width in dwords; stride_dw = row stride in
// dwords; LDS padding: interval = 2^pi eight-byte units, amount = code+1 dwords.
DEVINL void tdm_load_2d(unsigned lds_byte_addr, const void* gsrc,
                        unsigned tile_w_dw, unsigned tile_h,
                        unsigned long long stride_dw,
                        unsigned pad_interval_code, unsigned pad_amount_code,
                        bool pad_en) {
  unsigned long long ga = (unsigned long long)(uintptr_t)gsrc;
  u32x4 g0;
  g0.x = 1u;                                        // count=1 valid descriptor
  g0.y = lds_byte_addr;                             // lds_addr
  g0.z = (unsigned)ga;                              // global_addr[31:0]
  g0.w = (unsigned)(ga >> 32) | (2u << 30);         // global_addr[56:32] | type=2
  i32x8 g1;
  g1[0] = (int)((2u << 16) |                        // data_size = 4B
                (pad_en ? (1u << 20) : 0u) |        // pad_enable
                (pad_interval_code << 22) |
                (pad_amount_code << 25));
  g1[1] = (int)((tile_w_dw & 0xFFFFu) << 16);                         // tensor_dim0 lo
  g1[2] = (int)(((tile_w_dw >> 16) & 0xFFFFu) |
                ((tile_h & 0xFFFFu) << 16));                          // td0 hi | td1 lo
  g1[3] = (int)(((tile_h >> 16) & 0xFFFFu) |
                ((tile_w_dw & 0xFFFFu) << 16));                       // td1 hi | tile_dim0
  g1[4] = (int)(tile_h & 0xFFFFu);                                    // tile_dim1 (tile_dim2=0)
  g1[5] = (int)(unsigned)(stride_dw & 0xFFFFFFFFull);                 // dim0_stride lo
  g1[6] = (int)(unsigned)((stride_dw >> 32) & 0xFFFFull);             // dim0_stride hi
  g1[7] = 0;
  i32x4 gz = {0, 0, 0, 0};
#if __clang_major__ >= 23
  i32x8 gz8 = {0, 0, 0, 0, 0, 0, 0, 0};
  __builtin_amdgcn_tensor_load_to_lds(g0, g1, gz, gz, gz8, 0);
#else
  __builtin_amdgcn_tensor_load_to_lds(g0, g1, gz, gz, 0);
#endif
}
#endif  // HAVE_TDM

// ---------------------------------------------------------------------------
// f32 -> bf16 bulk convert (4 elements / thread)
__global__ void csa_cvt(const float* __restrict__ in, u16* __restrict__ out, int n4) {
  int i = blockIdx.x * blockDim.x + threadIdx.x;
  if (i >= n4) return;
  const float4 f = ((const float4*)in)[i];
  uint2 o;
  o.x = (unsigned)f2bf(f.x) | ((unsigned)f2bf(f.y) << 16);
  o.y = (unsigned)f2bf(f.z) | ((unsigned)f2bf(f.w) << 16);
  ((uint2*)out)[i] = o;
}

// f32 (K x N) -> bf16 transposed (N x K): Wt[n][k] = W[k][n]
__global__ void csa_cvtT(const float* __restrict__ W, u16* __restrict__ Wt, int K, int N) {
  long long id = (long long)blockIdx.x * 256 + threadIdx.x;
  if (id >= (long long)K * N) return;
  int k = (int)(id % K);
  int n = (int)(id / K);
  Wt[id] = f2bf(W[(size_t)k * N + n]);
}

// ---------------------------------------------------------------------------
// C[M,N] = A[M,K](bf16) * Bt[N,K]^T(bf16) + bias[N], tile 128x128x32, 8 waves.
// TDM path: double-buffered LDS; next tile's DMA overlaps current WMMAs.
template <bool OUT_BF16>
__global__ __launch_bounds__(256) void csa_gemm(const u16* __restrict__ A,
                                                const u16* __restrict__ Bt,
                                                const float* __restrict__ bias,
                                                void* __restrict__ out,
                                                int M, int N, int K) {
  constexpr int NBUF = HAVE_TDM ? 2 : 1;
  // row stride 40 halves = 20 dwords: TDM pads 4 dwords after every 16 dwords
  __shared__ __align__(16) u16 As[NBUF][128][40];
  __shared__ __align__(16) u16 Bs[NBUF][128][40];   // N-major: Bs[n][k]

  const int tid = threadIdx.x;
  const int lane = tid & 31, g = lane >> 4, lr = lane & 15;
  const int wave = tid >> 5;
  const int wr = wave >> 1, wc = wave & 1;          // 4x2 wave grid: 32x64 each
  const int bm = blockIdx.y * 128, bn = blockIdx.x * 128;

  v8f acc[2][4] = {};

#if HAVE_TDM
  // prologue: DMA tile 0
  if (wave == 0) {
    tdm_load_2d((unsigned)(uintptr_t)&As[0][0][0], A + (size_t)bm * K,
                16u, 128u, (unsigned long long)(K / 2), 3u, 3u, true);
    tdm_load_2d((unsigned)(uintptr_t)&Bs[0][0][0], Bt + (size_t)bn * K,
                16u, 128u, (unsigned long long)(K / 2), 3u, 3u, true);
    __builtin_amdgcn_s_wait_tensorcnt(0);
  }
  __syncthreads();
  int cur = 0;
  for (int k0 = 0; k0 < K; k0 += 32) {
    const int nxt = cur ^ 1;
    if (wave == 0 && k0 + 32 < K) {   // async prefetch of next K-tile
      tdm_load_2d((unsigned)(uintptr_t)&As[nxt][0][0], A + (size_t)bm * K + k0 + 32,
                  16u, 128u, (unsigned long long)(K / 2), 3u, 3u, true);
      tdm_load_2d((unsigned)(uintptr_t)&Bs[nxt][0][0], Bt + (size_t)bn * K + k0 + 32,
                  16u, 128u, (unsigned long long)(K / 2), 3u, 3u, true);
    }
    v16bf af[2], bf[4];
    #pragma unroll
    for (int mt = 0; mt < 2; ++mt) af[mt] = fragA(&As[cur][wr*32 + mt*16 + lr][0], 0, g);
    #pragma unroll
    for (int nt = 0; nt < 4; ++nt) bf[nt] = fragB(&Bs[cur][wc*64 + nt*16 + lr][0], 0, g);
    #pragma unroll
    for (int mt = 0; mt < 2; ++mt)
      #pragma unroll
      for (int nt = 0; nt < 4; ++nt)
        acc[mt][nt] = wmma_bf16(af[mt], bf[nt], acc[mt][nt]);
    if (wave == 0) __builtin_amdgcn_s_wait_tensorcnt(0);
    __syncthreads();
    cur = nxt;
  }
#else
  for (int k0 = 0; k0 < K; k0 += 32) {
    {
      const int ldr = tid >> 1, lseg = (tid & 1) * 16;
      const u16* ga = A  + (size_t)(bm + ldr) * K + k0 + lseg;
      const u16* gb = Bt + (size_t)(bn + ldr) * K + k0 + lseg;
      *(v8u*)&As[0][ldr][lseg]     = *(const v8u*)(ga);
      *(v8u*)&As[0][ldr][lseg + 8] = *(const v8u*)(ga + 8);
      *(v8u*)&Bs[0][ldr][lseg]     = *(const v8u*)(gb);
      *(v8u*)&Bs[0][ldr][lseg + 8] = *(const v8u*)(gb + 8);
      if (k0 + 32 < K) {
        __builtin_prefetch(ga + 32, 0, 0);
        __builtin_prefetch(gb + 32, 0, 0);
      }
    }
    __syncthreads();
    v16bf af[2], bf[4];
    #pragma unroll
    for (int mt = 0; mt < 2; ++mt) af[mt] = fragA(&As[0][wr*32 + mt*16 + lr][0], 0, g);
    #pragma unroll
    for (int nt = 0; nt < 4; ++nt) bf[nt] = fragB(&Bs[0][wc*64 + nt*16 + lr][0], 0, g);
    #pragma unroll
    for (int mt = 0; mt < 2; ++mt)
      #pragma unroll
      for (int nt = 0; nt < 4; ++nt)
        acc[mt][nt] = wmma_bf16(af[mt], bf[nt], acc[mt][nt]);
    __syncthreads();
  }
#endif

  #pragma unroll
  for (int mt = 0; mt < 2; ++mt) {
    #pragma unroll
    for (int nt = 0; nt < 4; ++nt) {
      const int col = bn + wc*64 + nt*16 + lr;
      const float bv = bias[col];
      #pragma unroll
      for (int e = 0; e < 8; ++e) {
        const int row = bm + wr*32 + mt*16 + e + 8*g;   // C/D layout: vgpr e, group g
        const float val = acc[mt][nt][e] + bv;
        if (OUT_BF16) ((u16*)out)[(size_t)row * N + col] = f2bf(val);
        else          ((float*)out)[(size_t)row * N + col] = val;
      }
    }
  }
}

// ---------------------------------------------------------------------------
// Flash attention: one block = 64 queries of one (b,h); loops over causal key
// blocks. TDM path: K tile for jb+1 prefetched while computing jb.
__global__ __launch_bounds__(256) void csa_attn(const u16* __restrict__ qkv,
                                                u16* __restrict__ Y) {
  constexpr int KBUF = HAVE_TDM ? 2 : 1;
  // row stride 72 halves = 36 dwords: TDM pads 4 dwords after every 32 dwords
  __shared__ __align__(16) u16 Qs[64][72];
  __shared__ __align__(16) u16 Ks[KBUF][64][72];
  __shared__ __align__(16) u16 VTs[64][72];   // transposed: VTs[d][key]
  __shared__ __align__(16) float Sbuf[64][68];
  __shared__ __align__(16) u16 Pbuf[64][72];
  __shared__ float mrow[64], lsum[64], arow[64];

  const int tid = threadIdx.x;
  const int lane = tid & 31, g = lane >> 4, lr = lane & 15;
  const int wave = tid >> 5;
  const int sr = wave >> 1, sc = wave & 1;    // wave tile: rows sr*16, cols sc*32(+16)
  const int bh = blockIdx.x, qb = blockIdx.y;
  const int b = bh / HH, h = bh % HH;

  const int lrw = tid >> 2;                   // 0..63 (tile row for loads)
  const int lsg = (tid & 3) * 16;             // 0,16,32,48

#if HAVE_TDM
  if (wave == 0) {
    const u16* qsrc = qkv + ((size_t)(b * TT + qb * 64)) * C3 + h * HD;
    tdm_load_2d((unsigned)(uintptr_t)&Qs[0][0], qsrc,
                32u, 64u, (unsigned long long)(C3 / 2), 4u, 3u, true);
    const u16* ksrc = qkv + ((size_t)(b * TT)) * C3 + CC + h * HD;   // jb = 0
    tdm_load_2d((unsigned)(uintptr_t)&Ks[0][0][0], ksrc,
                32u, 64u, (unsigned long long)(C3 / 2), 4u, 3u, true);
    __builtin_amdgcn_s_wait_tensorcnt(0);
  }
#else
  {
    const size_t src = ((size_t)(b * TT + qb * 64 + lrw)) * C3 + h * HD + lsg;
    *(v8u*)&Qs[lrw][lsg]     = *(const v8u*)(qkv + src);
    *(v8u*)&Qs[lrw][lsg + 8] = *(const v8u*)(qkv + src + 8);
  }
#endif
  if (tid < 64) { mrow[tid] = -3.0e38f; lsum[tid] = 0.f; }

  v8f o0 = {}, o1 = {};
  __syncthreads();

  int cur = 0;
  for (int jb = 0; jb <= qb; ++jb) {
    // ---- prefetch next K tile (TDM) / load K tile (fallback); stage V^T ----
#if HAVE_TDM
    if (wave == 0 && jb < qb) {
      const u16* ksrc = qkv + ((size_t)(b * TT + (jb + 1) * 64)) * C3 + CC + h * HD;
      tdm_load_2d((unsigned)(uintptr_t)&Ks[cur ^ 1][0][0], ksrc,
                  32u, 64u, (unsigned long long)(C3 / 2), 4u, 3u, true);
    }
    {
      const size_t vsrc = ((size_t)(b * TT + jb * 64 + lrw)) * C3 + 2 * CC + h * HD + lsg;
      const v8u v0 = *(const v8u*)(qkv + vsrc);
      const v8u v1 = *(const v8u*)(qkv + vsrc + 8);
      #pragma unroll
      for (int i = 0; i < 8; ++i) {
        VTs[lsg + i][lrw]     = v0[i];
        VTs[lsg + 8 + i][lrw] = v1[i];
      }
    }
#else
    {
      const size_t ksrc = ((size_t)(b * TT + jb * 64 + lrw)) * C3 + CC + h * HD + lsg;
      *(v8u*)&Ks[0][lrw][lsg]     = *(const v8u*)(qkv + ksrc);
      *(v8u*)&Ks[0][lrw][lsg + 8] = *(const v8u*)(qkv + ksrc + 8);
      const v8u v0 = *(const v8u*)(qkv + ksrc + CC);
      const v8u v1 = *(const v8u*)(qkv + ksrc + CC + 8);
      #pragma unroll
      for (int i = 0; i < 8; ++i) {
        VTs[lsg + i][lrw]     = v0[i];
        VTs[lsg + 8 + i][lrw] = v1[i];
      }
    }
#endif
    __syncthreads();

    // ---- S = Q * K^T  (64x64, K-dim = HD = 64) ----
    v8f s0 = {}, s1 = {};
    #pragma unroll
    for (int kk = 0; kk < 64; kk += 32) {
      v16bf aq = fragA(&Qs[sr*16 + lr][0], kk, g);
      v16bf b0 = fragB(&Ks[cur][sc*32 + lr][0], kk, g);
      v16bf b1 = fragB(&Ks[cur][sc*32 + 16 + lr][0], kk, g);
      s0 = wmma_bf16(aq, b0, s0);
      s1 = wmma_bf16(aq, b1, s1);
    }
    #pragma unroll
    for (int e = 0; e < 8; ++e) {
      const int r = sr*16 + e + 8*g;
      Sbuf[r][sc*32 + lr]      = s0[e];
      Sbuf[r][sc*32 + 16 + lr] = s1[e];
    }
    __syncthreads();

    // ---- online softmax: one thread per query row ----
    if (tid < 64) {
      const int r = tid;
      const int lim = (jb == qb) ? r : 63;      // causal mask inside diagonal block
      float rm = -3.0e38f;
      for (int c = 0; c <= lim; ++c) rm = fmaxf(rm, Sbuf[r][c] * 0.125f);
      const float mold = mrow[r];
      const float mnew = fmaxf(mold, rm);
      const float alpha = __expf(mold - mnew);
      float sum = 0.f;
      for (int c = 0; c < 64; ++c) {
        float p = (c <= lim) ? __expf(Sbuf[r][c] * 0.125f - mnew) : 0.f;
        sum += p;
        Pbuf[r][c] = f2bf(p);
      }
      lsum[r] = lsum[r] * alpha + sum;
      mrow[r] = mnew;
      arow[r] = alpha;
    }
    __syncthreads();

    // ---- rescale O accumulators, then O += P * V  (K-dim = 64 keys) ----
    #pragma unroll
    for (int e = 0; e < 8; ++e) {
      const float a = arow[sr*16 + e + 8*g];
      o0[e] *= a;
      o1[e] *= a;
    }
    #pragma unroll
    for (int kk = 0; kk < 64; kk += 32) {
      v16bf ap = fragA(&Pbuf[sr*16 + lr][0], kk, g);
      v16bf b0 = fragB(&VTs[sc*32 + lr][0], kk, g);
      v16bf b1 = fragB(&VTs[sc*32 + 16 + lr][0], kk, g);
      o0 = wmma_bf16(ap, b0, o0);
      o1 = wmma_bf16(ap, b1, o1);
    }
#if HAVE_TDM
    if (wave == 0) __builtin_amdgcn_s_wait_tensorcnt(0);
#endif
    __syncthreads();
#if HAVE_TDM
    cur ^= 1;
#endif
  }

  if (tid < 64) arow[tid] = 1.0f / lsum[tid];
  __syncthreads();

  #pragma unroll
  for (int e = 0; e < 8; ++e) {
    const int rloc = sr*16 + e + 8*g;
    const float il = arow[rloc];
    const int r = qb * 64 + rloc;
    const size_t orow = ((size_t)(b * TT + r)) * CC + h * HD;
    Y[orow + sc*32 + lr]      = f2bf(o0[e] * il);
    Y[orow + sc*32 + 16 + lr] = f2bf(o1[e] * il);
  }
}

// ---------------------------------------------------------------------------
extern "C" void kernel_launch(void* const* d_in, const int* in_sizes, int n_in,
                              void* d_out, int out_size, void* d_ws, size_t ws_size,
                              hipStream_t stream) {
  const float* x      = (const float*)d_in[0];
  const float* w_attn = (const float*)d_in[1];
  const float* b_attn = (const float*)d_in[2];
  const float* w_proj = (const float*)d_in[3];
  const float* b_proj = (const float*)d_in[4];

  const size_t MT = (size_t)BB * TT;                 // 8192 rows
  // workspace layout (bf16 buffers), ~68 MB total
  u16* Xbf = (u16*)d_ws;                             // MT x C
  u16* WaT = Xbf + MT * CC;                          // 3C x C (transposed w_attn)
  u16* WpT = WaT + (size_t)C3 * CC;                  // C x C  (transposed w_proj)
  u16* QKV = WpT + (size_t)CC * CC;                  // MT x 3C
  u16* Ybf = QKV + MT * C3;                          // MT x C

  {
    int n4 = (int)(MT * CC / 4);
    csa_cvt<<<(n4 + 255) / 256, 256, 0, stream>>>(x, Xbf, n4);
  }
  csa_cvtT<<<((CC * C3) + 255) / 256, 256, 0, stream>>>(w_attn, WaT, CC, C3);
  csa_cvtT<<<((CC * CC) + 255) / 256, 256, 0, stream>>>(w_proj, WpT, CC, CC);

  // QKV = X @ W_attn + b_attn  -> bf16
  csa_gemm<true><<<dim3(C3 / 128, (int)(MT / 128)), 256, 0, stream>>>(
      Xbf, WaT, b_attn, (void*)QKV, (int)MT, C3, CC);

  // flash attention -> Ybf
  csa_attn<<<dim3(BB * HH, TT / 64), 256, 0, stream>>>(QKV, Ybf);

  // out = Y @ W_proj + b_proj -> f32
  csa_gemm<false><<<dim3(CC / 128, (int)(MT / 128)), 256, 0, stream>>>(
      Ybf, WpT, b_proj, d_out, (int)MT, CC, CC);

  (void)in_sizes; (void)n_in; (void)out_size; (void)ws_size;
}
